// FrequencyLinearLayer_60215441490562
// MI455X (gfx1250) — matrix operand order repaired and checked by reading, not compiled, not run on hardware
//
#include <hip/hip_runtime.h>

// Problem sizes (fixed by the reference): x is (4,2048,1024) f32, W is 1024x1024 complex.
#define DSZ   1024
#define ROWS  8192            // B*N = 4*2048
#define PI2   6.283185307179586f

typedef __attribute__((ext_vector_type(2)))  float     v2f;
typedef __attribute__((ext_vector_type(4)))  float     v4f;
typedef __attribute__((ext_vector_type(8)))  float     v8f;
typedef __attribute__((ext_vector_type(4)))  _Float16  v4h;
typedef __attribute__((ext_vector_type(8)))  _Float16  v8h;
typedef __attribute__((ext_vector_type(16))) _Float16  v16h;

// gcc-style int4 vector: the pointee type the async-LDS builtin expects.
typedef int v4i_vec __attribute__((vector_size(4 * sizeof(int))));

__device__ __forceinline__ v16h cat8(v8h a, v8h b) {
    return __builtin_shufflevector(a, b, 0,1,2,3,4,5,6,7,8,9,10,11,12,13,14,15);
}

// ---------------------------------------------------------------------------
// CDNA5 async global->LDS staging (ASYNCcnt-tracked), guarded so the file
// compiles on toolchains where the builtins are absent (VGPR-copy fallback).
// Probe-derived signature: (as1 int4*, as3 int4*, imm offset, imm cpol).
// ---------------------------------------------------------------------------
#if __has_builtin(__builtin_amdgcn_global_load_async_to_lds_b128) && \
    __has_builtin(__builtin_amdgcn_s_wait_asynccnt)
#define USE_ASYNC_LDS 1
#else
#define USE_ASYNC_LDS 0
#endif

__device__ __forceinline__ void copy16_to_lds(const _Float16* g, _Float16* l) {
#if USE_ASYNC_LDS
    __builtin_amdgcn_global_load_async_to_lds_b128(
        (__attribute__((address_space(1))) v4i_vec*)g,
        (__attribute__((address_space(3))) v4i_vec*)l, 0, 0);
#else
    *(uint4*)l = *(const uint4*)g;
#endif
}

__device__ __forceinline__ void wait_staged() {
#if USE_ASYNC_LDS
    __builtin_amdgcn_s_wait_asynccnt(0);   // each wave waits for its own copies
#endif
    __syncthreads();                       // then all waves' data is visible
}

// ---------------------------------------------------------------------------
// Twiddle matrices: C[a,b] = cos(2*pi*a*b/D), S[a,b] = sin(2*pi*a*b/D).
// One symmetric pair serves both DFT stages. (a*b mod D) keeps angles exact.
// ---------------------------------------------------------------------------
__global__ void __launch_bounds__(256)
twiddle_mats_kernel(float* __restrict__ C, float* __restrict__ S) {
    const int idx = blockIdx.x * 256 + threadIdx.x;   // over D*D
    const int a = idx >> 10;
    const int b = idx & (DSZ - 1);
    const int t = (a * b) & (DSZ - 1);
    const float ang = (PI2 / (float)DSZ) * (float)t;
    C[idx] = cosf(ang);
    S[idx] = sinf(ang);
}

// ---------------------------------------------------------------------------
// Split-f16 conversion: src f32 -> hi/lo f16 planes (hi = rn(src),
// lo = rn(src - hi)); used once for X and for M^T, removing all split VALU
// work from the GEMM hot loop and halving staged bytes.
// ---------------------------------------------------------------------------
__device__ __forceinline__ void split4(v4f v, v4h& h, v4h& l) {
    h.x = (_Float16)v.x; h.y = (_Float16)v.y; h.z = (_Float16)v.z; h.w = (_Float16)v.w;
    l.x = (_Float16)(v.x - (float)h.x); l.y = (_Float16)(v.y - (float)h.y);
    l.z = (_Float16)(v.z - (float)h.z); l.w = (_Float16)(v.w - (float)h.w);
}

__global__ void __launch_bounds__(256)
split_convert_kernel(const float* __restrict__ src, _Float16* __restrict__ hi,
                     _Float16* __restrict__ lo, int n4) {
    const int idx = blockIdx.x * 256 + threadIdx.x;
    if (idx < n4) {
        v4f v = ((const v4f*)src)[idx];
        v4h h, l; split4(v, h, l);
        ((v4h*)hi)[idx] = h;
        ((v4h*)lo)[idx] = l;
    }
}

// ---------------------------------------------------------------------------
// f32 WMMA GEMM (V_WMMA_F32_16X16X4_F32), 1024^3, full-f32 weight-prep chain:
//   Out (= or +=) scale * (A @ B)     [optionally stored transposed]
// B tile staged TRANSPOSED in LDS so each B-fragment K-pair is one aligned
// ds_load_b64 (no v_mov repacking). Prefetch of next tiles -> global_prefetch.
// ---------------------------------------------------------------------------
#define BM 128
#define BN 64
#define BK 32
#define XS_ST 36   // f32 stride: step 36 dwords mod 64 banks -> conflict-free
#define WT_ST 34   // transposed-B stride (even -> b64-aligned pairs)

__global__ void __launch_bounds__(256)
gemm_f32_wmma(const float* __restrict__ A, const float* __restrict__ B,
              float* __restrict__ Out, float scale, int accumulate, int transpose_out) {
    __shared__ float Xs[BM * XS_ST];    // 18432 B
    __shared__ float Wst[BN * WT_ST];   //  8704 B, [n][k]

    const int tid   = threadIdx.x;
    const int lane  = tid & 31;
    const int wave  = tid >> 5;
    const int half  = lane >> 4;
    const int l15   = lane & 15;
    const int waveM = wave & 3;
    const int waveN = wave >> 2;
    const int blockM = blockIdx.x * BM;
    const int blockN = blockIdx.y * BN;

    v8f acc[2][2];
#pragma unroll
    for (int i = 0; i < 2; ++i)
#pragma unroll
        for (int j = 0; j < 2; ++j) acc[i][j] = (v8f){0.f,0.f,0.f,0.f,0.f,0.f,0.f,0.f};

    for (int k0 = 0; k0 < DSZ; k0 += BK) {
        if (k0 + BK < DSZ) {                       // next-tile prefetch
            if (tid < 128)
                __builtin_prefetch(A + (size_t)(blockM + tid) * DSZ + k0 + BK, 0, 1);
            else if (tid < 192) {
                const int t = tid - 128;
                __builtin_prefetch(B + (size_t)(k0 + BK + (t >> 1)) * DSZ + blockN + (t & 1) * 32, 0, 1);
            }
        }
#pragma unroll
        for (int i = 0; i < 4; ++i) {              // A tile: 128x32 = 1024 float4
            const int idx = tid + i * 256;
            const int row = idx >> 3;
            const int c4  = idx & 7;
            v4f v = *(const v4f*)(A + (size_t)(blockM + row) * DSZ + k0 + c4 * 4);
            float* dst = &Xs[row * XS_ST + c4 * 4];
            dst[0] = v.x; dst[1] = v.y; dst[2] = v.z; dst[3] = v.w;
        }
#pragma unroll
        for (int i = 0; i < 2; ++i) {              // B tile: 32x64, scaled + transposed
            const int idx = tid + i * 256;
            const int row = idx >> 4;              // k row
            const int c4  = idx & 15;              // n quad
            v4f v = *(const v4f*)(B + (size_t)(k0 + row) * DSZ + blockN + c4 * 4);
            const int nb = c4 * 4;
            Wst[(nb + 0) * WT_ST + row] = v.x * scale;
            Wst[(nb + 1) * WT_ST + row] = v.y * scale;
            Wst[(nb + 2) * WT_ST + row] = v.z * scale;
            Wst[(nb + 3) * WT_ST + row] = v.w * scale;
        }
        __syncthreads();

#pragma unroll
        for (int kk = 0; kk < BK; kk += 4) {
            const int ke = kk + 2 * half;          // lane half picks K pair {ke,ke+1}
            v2f a[2], b[2];
#pragma unroll
            for (int tm = 0; tm < 2; ++tm) {
                const float* p = &Xs[(waveM * 32 + tm * 16 + l15) * XS_ST + ke];
                a[tm].x = p[0]; a[tm].y = p[1];
            }
#pragma unroll
            for (int tn = 0; tn < 2; ++tn) {
                const float* p = &Wst[(waveN * 32 + tn * 16 + l15) * WT_ST + ke];
                b[tn].x = p[0]; b[tn].y = p[1];
            }
#pragma unroll
            for (int tm = 0; tm < 2; ++tm)
#pragma unroll
                for (int tn = 0; tn < 2; ++tn)
                    acc[tm][tn] = __builtin_amdgcn_wmma_f32_16x16x4_f32(
                        false, a[tm], false, b[tn], (short)0, acc[tm][tn], false, false);
        }
        __syncthreads();
    }

    // Epilogue. C layout: VGPR r -> row r (lanes 0-15) / r+8 (lanes 16-31).
    if (transpose_out) {
#pragma unroll
        for (int tm = 0; tm < 2; ++tm)
#pragma unroll
            for (int tn = 0; tn < 2; ++tn) {
                const int col   = blockN + waveN * 32 + tn * 16 + l15;
                const int rbase = blockM + waveM * 32 + tm * 16 + 8 * half;
                float* p = Out + (size_t)col * DSZ + rbase;
                v4f lo = __builtin_shufflevector(acc[tm][tn], acc[tm][tn], 0,1,2,3);
                v4f hi = __builtin_shufflevector(acc[tm][tn], acc[tm][tn], 4,5,6,7);
                if (accumulate) { lo += *(v4f*)p; hi += *(v4f*)(p + 4); }
                *(v4f*)p = lo; *(v4f*)(p + 4) = hi;
            }
    } else {
#pragma unroll
        for (int tm = 0; tm < 2; ++tm)
#pragma unroll
            for (int tn = 0; tn < 2; ++tn) {
                const int col   = blockN + waveN * 32 + tn * 16 + l15;
                const int rbase = blockM + waveM * 32 + tm * 16 + 8 * half;
#pragma unroll
                for (int r = 0; r < 8; ++r) {
                    float* p = Out + (size_t)(rbase + r) * DSZ + col;
                    float v = acc[tm][tn][r];
                    if (accumulate) v += *p;
                    *p = v;
                }
            }
    }
}

// ---------------------------------------------------------------------------
// Final GEMM: Y(8192x1024) = X @ M + bias via split-f16 on
// V_WMMA_F32_16X16X32_F16 (xh*Mh + xh*Ml + xl*Mh, f32 accumulate, ~2^-22 err).
// Inputs are prebuilt f16 hi/lo planes (X row-major, M transposed), so LDS
// staging is pure b128 async copies, DOUBLE-BUFFERED:
//   issue async copies for tile k+1 -> 12 WMMAs on tile k -> wait_async+barrier
// A-fragments: runs of 8 halfwords at K = 8*half and 16+8*half (ISA layout).
// B-fragments: runs of 8 halfwords at K = 8*half and 16+8*half (matches A).
// ---------------------------------------------------------------------------
#define FBM 128
#define FBN 64
#define FBK 32
#define FA_ST 40   // f16 stride (80B = 20 dwords): conflict-free, 16B aligned
#define FB_ST 40

__device__ __forceinline__ void stage_tiles(
    const _Float16* __restrict__ Xhi, const _Float16* __restrict__ Xlo,
    const _Float16* __restrict__ Mthi, const _Float16* __restrict__ Mtlo,
    _Float16* aHi, _Float16* aLo, _Float16* bHi, _Float16* bLo,
    int blockM, int blockN, int k0, int tid) {
#pragma unroll
    for (int i = 0; i < 2; ++i) {                  // A planes: 128x32 f16, 2 b128/thread/plane
        const int idx = tid + i * 256;
        const int row = idx >> 2;
        const int c8  = (idx & 3) * 8;
        const size_t g = (size_t)(blockM + row) * DSZ + k0 + c8;
        copy16_to_lds(Xhi + g, &aHi[row * FA_ST + c8]);
        copy16_to_lds(Xlo + g, &aLo[row * FA_ST + c8]);
    }
    {                                              // B planes: 64x32 f16, 1 b128/thread/plane
        const int row = tid >> 2;
        const int c8  = (tid & 3) * 8;
        const size_t g = (size_t)(blockN + row) * DSZ + k0 + c8;
        copy16_to_lds(Mthi + g, &bHi[row * FB_ST + c8]);
        copy16_to_lds(Mtlo + g, &bLo[row * FB_ST + c8]);
    }
}

__global__ void __launch_bounds__(256)
gemm_f16split_wmma(const _Float16* __restrict__ Xhi, const _Float16* __restrict__ Xlo,
                   const _Float16* __restrict__ Mthi, const _Float16* __restrict__ Mtlo,
                   const float* __restrict__ bias, float* __restrict__ Y) {
    __shared__ _Float16 Ahi[2][FBM * FA_ST];   // 2 x 10240 B
    __shared__ _Float16 Alo[2][FBM * FA_ST];   // 2 x 10240 B
    __shared__ _Float16 Bhi[2][FBN * FB_ST];   // 2 x  5120 B
    __shared__ _Float16 Blo[2][FBN * FB_ST];   // 2 x  5120 B  -> 61440 B total

    const int tid   = threadIdx.x;
    const int lane  = tid & 31;
    const int wave  = tid >> 5;
    const int half  = lane >> 4;
    const int l15   = lane & 15;
    const int waveM = wave & 3;
    const int waveN = wave >> 2;
    const int blockM = blockIdx.x * FBM;
    const int blockN = blockIdx.y * FBN;

    v8f acc[2][2];
#pragma unroll
    for (int i = 0; i < 2; ++i)
#pragma unroll
        for (int j = 0; j < 2; ++j) acc[i][j] = (v8f){0.f,0.f,0.f,0.f,0.f,0.f,0.f,0.f};

    stage_tiles(Xhi, Xlo, Mthi, Mtlo, Ahi[0], Alo[0], Bhi[0], Blo[0],
                blockM, blockN, 0, tid);
    wait_staged();

    int buf = 0;
    for (int k0 = 0; k0 < DSZ; k0 += FBK) {
        const int nxt = k0 + FBK;
        if (nxt < DSZ)                             // issue async copies for tile k+1
            stage_tiles(Xhi, Xlo, Mthi, Mtlo, Ahi[buf ^ 1], Alo[buf ^ 1],
                        Bhi[buf ^ 1], Blo[buf ^ 1], blockM, blockN, nxt, tid);

        v16h ah[2], al[2], bh[2], bl[2];
#pragma unroll
        for (int tm = 0; tm < 2; ++tm) {
            const int base = (waveM * 32 + tm * 16 + l15) * FA_ST;
            ah[tm] = cat8(*(const v8h*)&Ahi[buf][base + 8 * half],
                          *(const v8h*)&Ahi[buf][base + 16 + 8 * half]);
            al[tm] = cat8(*(const v8h*)&Alo[buf][base + 8 * half],
                          *(const v8h*)&Alo[buf][base + 16 + 8 * half]);
        }
#pragma unroll
        for (int tn = 0; tn < 2; ++tn) {
            const int base = (waveN * 32 + tn * 16 + l15) * FB_ST + 8 * half;
            bh[tn] = cat8(*(const v8h*)&Bhi[buf][base], *(const v8h*)&Bhi[buf][base + 16]);
            bl[tn] = cat8(*(const v8h*)&Blo[buf][base], *(const v8h*)&Blo[buf][base + 16]);
        }
#pragma unroll
        for (int tm = 0; tm < 2; ++tm)
#pragma unroll
            for (int tn = 0; tn < 2; ++tn) {
                acc[tm][tn] = __builtin_amdgcn_wmma_f32_16x16x32_f16(
                    false, ah[tm], false, bh[tn], (short)0, acc[tm][tn], false, false);
                acc[tm][tn] = __builtin_amdgcn_wmma_f32_16x16x32_f16(
                    false, ah[tm], false, bl[tn], (short)0, acc[tm][tn], false, false);
                acc[tm][tn] = __builtin_amdgcn_wmma_f32_16x16x32_f16(
                    false, al[tm], false, bh[tn], (short)0, acc[tm][tn], false, false);
            }

        wait_staged();                             // next buffer landed; buf reusable
        buf ^= 1;
    }

    // Epilogue: bias add + f32 store.
#pragma unroll
    for (int tm = 0; tm < 2; ++tm)
#pragma unroll
        for (int tn = 0; tn < 2; ++tn) {
            const int col   = blockN + waveN * 32 + tn * 16 + l15;
            const float bv  = bias[col];
            const int rbase = blockM + waveM * 32 + tm * 16 + 8 * half;
#pragma unroll
            for (int r = 0; r < 8; ++r)
                Y[(size_t)(rbase + r) * DSZ + col] = acc[tm][tn][r] + bv;
        }
}

// ---------------------------------------------------------------------------
// Launch chain (all matrix work on WMMA):
//   twiddles C,S
//   Tr =  (1/D) wr@C ;  Tr += (-1/D) wi@S      (row-wise IFFT of W, real)
//   Ti =  (1/D) wr@S ;  Ti +=  (1/D) wi@C      (row-wise IFFT of W, imag)
//   Mt =  (C@Tr)^T   ;  Mt += (S@Ti)^T         (real col-DFT -> M, stored T)
//   Mt -> Mthi/Mtlo planes ; x -> Xhi/Xlo planes
//   Y  =  X@M + bias                            (split-f16 WMMA, async+dbuf)
// Workspace (f32 units), overlaid since C/S/Tr/Ti die before Xhi/Xlo exist:
//   Mt[N2] | Mthi[N2/2] | Mtlo[N2/2] | R: {C,S,Tr,Ti}[4*N2] then {Xhi,Xlo}[8*N2]
//   total 10*N2 f32 = 40 MB (L2-resident).
// ---------------------------------------------------------------------------
extern "C" void kernel_launch(void* const* d_in, const int* in_sizes, int n_in,
                              void* d_out, int out_size, void* d_ws, size_t ws_size,
                              hipStream_t stream) {
    const float* x    = (const float*)d_in[0];
    const float* wr   = (const float*)d_in[1];
    const float* wi   = (const float*)d_in[2];
    const float* bias = (const float*)d_in[3];
    float* out = (float*)d_out;

    const size_t N2 = (size_t)DSZ * DSZ;
    float*     Mt   = (float*)d_ws;
    _Float16*  Mthi = (_Float16*)(Mt + N2);
    _Float16*  Mtlo = Mthi + N2;
    float*     R    = Mt + 2 * N2;
    float*     Cm   = R;
    float*     Sm   = R + N2;
    float*     Tr   = R + 2 * N2;
    float*     Ti   = R + 3 * N2;
    _Float16*  Xhi  = (_Float16*)R;            // reuses C..Ti after they die
    _Float16*  Xlo  = Xhi + (size_t)ROWS * DSZ;

    twiddle_mats_kernel<<<dim3((DSZ * DSZ) / 256), dim3(256), 0, stream>>>(Cm, Sm);

    const float inv = 1.0f / (float)DSZ;
    dim3 g32(DSZ / BM, DSZ / BN);   // 8 x 16
    gemm_f32_wmma<<<g32, dim3(256), 0, stream>>>(wr, Cm, Tr,  inv, 0, 0);
    gemm_f32_wmma<<<g32, dim3(256), 0, stream>>>(wi, Sm, Tr, -inv, 1, 0);
    gemm_f32_wmma<<<g32, dim3(256), 0, stream>>>(wr, Sm, Ti,  inv, 0, 0);
    gemm_f32_wmma<<<g32, dim3(256), 0, stream>>>(wi, Cm, Ti,  inv, 1, 0);
    gemm_f32_wmma<<<g32, dim3(256), 0, stream>>>(Cm, Tr, Mt, 1.0f, 0, 1);
    gemm_f32_wmma<<<g32, dim3(256), 0, stream>>>(Sm, Ti, Mt, 1.0f, 1, 1);

    split_convert_kernel<<<dim3((int)(N2 / 4 / 256)), dim3(256), 0, stream>>>(
        Mt, Mthi, Mtlo, (int)(N2 / 4));
    split_convert_kernel<<<dim3((int)((size_t)ROWS * DSZ / 4 / 256)), dim3(256), 0, stream>>>(
        x, Xhi, Xlo, (int)((size_t)ROWS * DSZ / 4));

    dim3 gf(ROWS / FBM, DSZ / FBN); // 64 x 16
    gemm_f16split_wmma<<<gf, dim3(256), 0, stream>>>(Xhi, Xlo, Mthi, Mtlo, bias, out);
}
